// PointNetConv_64707977282170
// MI455X (gfx1250) — compile-verified, matrix operand order checked by reading, not run on hardware
//
#include <hip/hip_runtime.h>
#include <hip/hip_bf16.h>

typedef __bf16 bf16_t;
typedef __attribute__((ext_vector_type(16))) __bf16 v16bf;
typedef __attribute__((ext_vector_type(8)))  __bf16 v8bf;
typedef __attribute__((ext_vector_type(8)))  float  v8f;

#define N_NODES 50000
#define N_EDGES 1600000
#define CIN  128
#define COUT 128
#define K1PAD 160   // 131 (CIN+CPOS) padded to 5*32 for bf16 WMMA K-chunks
#define K2   128    // 4*32
#define MTILE 32    // edges per workgroup iteration (two 16-row WMMA M-tiles)

// LDS row strides in bf16 elements; *2B are 16B-aligned and skewed vs the
// 64 x 4B banks (352B -> 24-bank shift/row, 272B -> 4-bank shift/row).
#define STRA 176
#define STR1 136

// ---------------------------------------------------------------------------
// Pack W1/W2/Wr into bf16, transposed (column-major => each lane's 16-wide
// K-fragment is contiguous), with W1's K padded 131 -> 160 with zeros.
// ---------------------------------------------------------------------------
__global__ void pack_weights_kernel(const float* __restrict__ W1,
                                    const float* __restrict__ W2,
                                    const float* __restrict__ Wr,
                                    bf16_t* __restrict__ Wp1,
                                    bf16_t* __restrict__ Wp2,
                                    bf16_t* __restrict__ Wrp) {
  int t = blockIdx.x * blockDim.x + threadIdx.x;
  if (t < COUT * K1PAD) {
    int n = t / K1PAD, k = t % K1PAD;
    Wp1[n * K1PAD + k] = (bf16_t)((k < 131) ? W1[k * COUT + n] : 0.0f);
  }
  if (t < COUT * K2) {
    int n = t / K2, k = t % K2;
    Wp2[n * K2 + k] = (bf16_t)W2[k * COUT + n];
    Wrp[n * K2 + k] = (bf16_t)Wr[k * COUT + n];
  }
}

// segment-max identity
__global__ void seg_init_kernel(float* __restrict__ out) {
  int i = blockIdx.x * blockDim.x + threadIdx.x;
  if (i < N_NODES * COUT) out[i] = -INFINITY;
}

__device__ __forceinline__ v8bf pack8(float4 a, float4 b) {
  v8bf r;
  r[0] = (bf16_t)a.x; r[1] = (bf16_t)a.y; r[2] = (bf16_t)a.z; r[3] = (bf16_t)a.w;
  r[4] = (bf16_t)b.x; r[5] = (bf16_t)b.y; r[6] = (bf16_t)b.z; r[7] = (bf16_t)b.w;
  return r;
}

// ---------------------------------------------------------------------------
// Edge MLP: per 32-edge tile, vectorized gather -> bf16 LDS -> WMMA L1 ->
// ReLU -> WMMA L2 -> atomic f32 max scatter. 8 waves; wave w owns output
// columns [16w, 16w+16). B fragments hoisted outside the persistent loop;
// A fragments prefetched as arrays so WMMAs issue back-to-back.
// ---------------------------------------------------------------------------
__global__ void __launch_bounds__(256)
edge_mlp_kernel(const float* __restrict__ x,
                const float* __restrict__ pos,
                const int*   __restrict__ eidx,
                const bf16_t* __restrict__ Wp1,
                const float* __restrict__ b1,
                const bf16_t* __restrict__ Wp2,
                const float* __restrict__ b2,
                float* __restrict__ seg) {
  __shared__ __attribute__((aligned(16))) bf16_t hA[MTILE * STRA];
  __shared__ __attribute__((aligned(16))) bf16_t h1[MTILE * STR1];
  __shared__ int srow[MTILE];
  __shared__ int scol[MTILE];

  const int t    = threadIdx.x;
  const int lane = t & 31;
  const int w    = t >> 5;        // wave id 0..7 -> N tile
  const int n16  = lane & 15;     // N (and A-row) index within tile
  const int hl   = lane >> 4;     // K-half within a 32-wide chunk
  const int ncol = w * 16 + n16;  // output column 0..127

  // Hoisted B fragments (bf16, 32 B contiguous per lane).
  v16bf b1f[5];
#pragma unroll
  for (int kc = 0; kc < 5; ++kc)
    b1f[kc] = *(const v16bf*)(Wp1 + ncol * K1PAD + kc * 32 + hl * 16);
  v16bf b2f[4];
#pragma unroll
  for (int kc = 0; kc < 4; ++kc)
    b2f[kc] = *(const v16bf*)(Wp2 + ncol * K2 + kc * 32 + hl * 16);
  const float b1v = b1[ncol];
  const float b2v = b2[ncol];

  const int ntiles = N_EDGES / MTILE;  // 50000, exact
  for (int tile = blockIdx.x; tile < ntiles; tile += gridDim.x) {
    const int base = tile * MTILE;
    if (t < MTILE) {
      srow[t] = eidx[base + t];             // source
      scol[t] = eidx[N_EDGES + base + t];   // destination
    }
    if (t == MTILE) {
      __builtin_prefetch(eidx + base + MTILE, 0, 0);            // next rows
      __builtin_prefetch(eidx + N_EDGES + base + MTILE, 0, 0);  // next cols
    }
    __syncthreads();

    // Gather x[row] (32 edges x 128 f32, L2-resident) -> bf16 LDS A tile.
    // Thread t: edge row m = t>>3, 16-col segment q = t&7 (two b128 stores).
    {
      const int m = t >> 3, q = t & 7;
      const float4* xr = (const float4*)(x + (long)srow[m] * CIN) + q * 4;
      float4 f0 = xr[0], f1 = xr[1], f2 = xr[2], f3 = xr[3];
      *(v8bf*)(hA + m * STRA + q * 16)     = pack8(f0, f1);
      *(v8bf*)(hA + m * STRA + q * 16 + 8) = pack8(f2, f3);
    }
    if (t < MTILE) {  // rel = pos[row]-pos[col] into k=128..130, pad to 160
      const int r = srow[t], c = scol[t];
#pragma unroll
      for (int cc = 0; cc < 3; ++cc)
        hA[t * STRA + 128 + cc] = (bf16_t)(pos[r * 3 + cc] - pos[c * 3 + cc]);
      for (int k = 131; k < K1PAD; ++k) hA[t * STRA + k] = (bf16_t)0.0f;
    }
    __syncthreads();

    // Layer 1 for both 16-edge sub-tiles: [16x160] @ [160x16] (bf16->f32)
#pragma unroll
    for (int sub = 0; sub < 2; ++sub) {
      const int arow = sub * 16 + n16;
      v16bf a[5];
#pragma unroll
      for (int kc = 0; kc < 5; ++kc)  // issue all loads, then all WMMAs
        a[kc] = *(const v16bf*)(hA + arow * STRA + kc * 32 + hl * 16);
      v8f c1 = {};
#pragma unroll
      for (int kc = 0; kc < 5; ++kc)
        c1 = __builtin_amdgcn_wmma_f32_16x16x32_bf16(false, a[kc], false,
                                                     b1f[kc], (short)0, c1,
                                                     false, false);
      // bias + ReLU -> bf16 LDS in A-matrix layout for layer 2
#pragma unroll
      for (int v = 0; v < 8; ++v) {
        float r = c1[v] + b1v;
        r = r > 0.0f ? r : 0.0f;
        h1[(sub * 16 + v + 8 * hl) * STR1 + ncol] = (bf16_t)r;
      }
    }
    __syncthreads();

    // Layer 2 for both sub-tiles: [16x128] @ [128x16] + atomic max scatter
#pragma unroll
    for (int sub = 0; sub < 2; ++sub) {
      const int arow = sub * 16 + n16;
      v16bf a[4];
#pragma unroll
      for (int kc = 0; kc < 4; ++kc)
        a[kc] = *(const v16bf*)(h1 + arow * STR1 + kc * 32 + hl * 16);
      v8f c2 = {};
#pragma unroll
      for (int kc = 0; kc < 4; ++kc)
        c2 = __builtin_amdgcn_wmma_f32_16x16x32_bf16(false, a[kc], false,
                                                     b2f[kc], (short)0, c2,
                                                     false, false);
#pragma unroll
      for (int v = 0; v < 8; ++v) {
        const int m = sub * 16 + v + 8 * hl;  // edge within tile
        float* addr = seg + (long)scol[m] * COUT + ncol;
        __hip_atomic_fetch_max(addr, c2[v] + b2v, __ATOMIC_RELAXED,
                               __HIP_MEMORY_SCOPE_AGENT);
      }
    }
    __syncthreads();  // protect LDS reuse next iteration
  }
}

// ---------------------------------------------------------------------------
// Finalize: out = where(out==-inf, 0, out) + x @ Wr + br, one 16-node tile
// per workgroup, WMMA for the GEMM.
// ---------------------------------------------------------------------------
__global__ void __launch_bounds__(256)
finalize_kernel(const float* __restrict__ x,
                const bf16_t* __restrict__ Wrp,
                const float* __restrict__ br,
                float* __restrict__ out) {
  __shared__ __attribute__((aligned(16))) bf16_t xs[16 * STR1];
  const int t    = threadIdx.x;
  const int lane = t & 31;
  const int w    = t >> 5;
  const int n16  = lane & 15;
  const int hl   = lane >> 4;
  const int ncol = w * 16 + n16;

  v16bf bfrag[4];
#pragma unroll
  for (int kc = 0; kc < 4; ++kc)
    bfrag[kc] = *(const v16bf*)(Wrp + ncol * K2 + kc * 32 + hl * 16);
  const float brv = br[ncol];

  const int row0 = blockIdx.x * 16;  // N_NODES/16 blocks, exact
  {
    const int m = t >> 4, l16 = t & 15;  // 16 threads per node row
    const float4* xr = (const float4*)(x + (long)(row0 + m) * CIN) + l16 * 2;
    float4 f0 = xr[0], f1 = xr[1];
    *(v8bf*)(xs + m * STR1 + l16 * 8) = pack8(f0, f1);
  }
  __syncthreads();

  v16bf a[4];
#pragma unroll
  for (int kc = 0; kc < 4; ++kc)
    a[kc] = *(const v16bf*)(xs + n16 * STR1 + kc * 32 + hl * 16);
  v8f c = {};
#pragma unroll
  for (int kc = 0; kc < 4; ++kc)
    c = __builtin_amdgcn_wmma_f32_16x16x32_bf16(false, a[kc], false, bfrag[kc],
                                                (short)0, c, false, false);
#pragma unroll
  for (int v = 0; v < 8; ++v) {
    const int node = row0 + v + 8 * hl;
    float* p = out + (long)node * COUT + ncol;
    float s = *p;                       // segment-max result
    if (s == -INFINITY) s = 0.0f;       // nodes with no incoming edges
    *p = s + c[v] + brv;
  }
}

// ---------------------------------------------------------------------------
extern "C" void kernel_launch(void* const* d_in, const int* in_sizes, int n_in,
                              void* d_out, int out_size, void* d_ws, size_t ws_size,
                              hipStream_t stream) {
  const float* x   = (const float*)d_in[0];
  const float* pos = (const float*)d_in[1];
  const int*   ei  = (const int*)d_in[2];
  const float* W1  = (const float*)d_in[3];
  const float* b1  = (const float*)d_in[4];
  const float* W2  = (const float*)d_in[5];
  const float* b2  = (const float*)d_in[6];
  const float* Wr  = (const float*)d_in[7];
  const float* br  = (const float*)d_in[8];
  float* out = (float*)d_out;

  // ws layout: packed bf16 weights, ~104 KB total.
  char* ws = (char*)d_ws;
  bf16_t* Wp1 = (bf16_t*)(ws);                           // 128*160*2 = 40960 B
  bf16_t* Wp2 = (bf16_t*)(ws + 40960);                   // 128*128*2 = 32768 B
  bf16_t* Wrp = (bf16_t*)(ws + 40960 + 32768);           // 32768 B

  pack_weights_kernel<<<(COUT * K1PAD + 255) / 256, 256, 0, stream>>>(
      W1, W2, Wr, Wp1, Wp2, Wrp);
  seg_init_kernel<<<(N_NODES * COUT + 255) / 256, 256, 0, stream>>>(out);
  edge_mlp_kernel<<<4096, 256, 0, stream>>>(x, pos, ei, Wp1, b1, Wp2, b2, out);
  finalize_kernel<<<N_NODES / 16, 256, 0, stream>>>(x, Wrp, br, out);
}
